// fgateLstmUnit_8735963480662
// MI455X (gfx1250) — matrix operand order repaired and checked by reading, not compiled, not run on hardware
//
#include <hip/hip_runtime.h>

#define HS   512
#define INW  512
#define FS   128
#define NB   128   // batch
#define SQ   512   // sequence

typedef __attribute__((ext_vector_type(16))) __bf16 v16bf;
typedef __attribute__((ext_vector_type(8)))  float  v8f;

union BF16Frag {
  v16bf v;
  uint4 q[2];
};

__device__ __forceinline__ unsigned short f32_to_bf16(float f) {
  union { float f; unsigned int u; } x; x.f = f;
  unsigned int u = x.u;
  u += 0x7FFFu + ((u >> 16) & 1u);   // round-to-nearest-even
  return (unsigned short)(u >> 16);
}

__device__ __forceinline__ float sigmoidf(float x) {
  return 1.0f / (1.0f + __expf(-x));
}

// ---------------------------------------------------------------------------
// Pre-swizzle weights into WMMA A-fragment order (bf16).
// Gates: A = [W_f ; U_f]^T tiles, 128 N-tiles x 32 K-tiles, 512 bf16/tile.
// Z:     A = Wz^T tiles,          64 N-tiles x  4 K-tiles, 512 bf16/tile.
// Element (L, i) of a tile: n = nt*16 + (L&15),
//                           k = kt*32 + ((i<8)? i : i+8) + 8*(L>>4)  (ISA A 16x32)
// ---------------------------------------------------------------------------
__global__ void prep_weights(const float* __restrict__ Wf,
                             const float* __restrict__ Uf,
                             const float* __restrict__ Wz,
                             unsigned short* __restrict__ WsG,
                             unsigned short* __restrict__ WsZ) {
  int e = blockIdx.x * blockDim.x + threadIdx.x;
  const int NG = 128 * 32 * 512;
  const int NZ = 64 * 4 * 512;
  if (e < NG) {
    int tile = e >> 9, pos = e & 511;
    int L = pos >> 4, i = pos & 15;
    int nt = tile >> 5, kt = tile & 31;
    int n = nt * 16 + (L & 15);
    int k = kt * 32 + ((i < 8) ? i : i + 8) + 8 * (L >> 4);
    float v = (k < INW) ? Wf[(size_t)k * (4 * HS) + n]
                        : Uf[(size_t)(k - INW) * (4 * HS) + n];
    WsG[e] = f32_to_bf16(v);
  } else if (e < NG + NZ) {
    int ez = e - NG;
    int tile = ez >> 9, pos = ez & 511;
    int L = pos >> 4, i = pos & 15;
    int nt = tile >> 2, kt = tile & 3;
    int n = nt * 16 + (L & 15);
    int k = kt * 32 + ((i < 8) ? i : i + 8) + 8 * (L >> 4);
    WsZ[ez] = f32_to_bf16(Wz[(size_t)k * (2 * HS) + n]);
  }
}

__global__ void init_state(float* __restrict__ h0, float* __restrict__ c,
                           unsigned* __restrict__ bar) {
  int e = blockIdx.x * blockDim.x + threadIdx.x;
  if (e < NB * HS) { h0[e] = 0.0f; c[e] = 0.0f; }
  if (e == 0) *bar = 0u;
}

// ---------------------------------------------------------------------------
// Grid-wide barrier: all 64 blocks are co-resident (64 << #WGPs on MI455X),
// so a counter barrier is safe. Release fence before signal, acquire fence
// after the spin; waves of a block share the WGP$ so thread0's invalidate
// covers the whole workgroup.
// ---------------------------------------------------------------------------
__device__ __forceinline__ void grid_barrier(unsigned* bar, unsigned target) {
  __syncthreads();
  if (threadIdx.x == 0) {
    __threadfence();
    __hip_atomic_fetch_add(bar, 1u, __ATOMIC_RELEASE, __HIP_MEMORY_SCOPE_AGENT);
    while (__hip_atomic_load(bar, __ATOMIC_ACQUIRE, __HIP_MEMORY_SCOPE_AGENT) < target) {
      __builtin_amdgcn_s_sleep(1);
    }
    __threadfence();
  }
  __syncthreads();
}

// ---------------------------------------------------------------------------
// Persistent recurrence kernel. Grid = (4 batch-tiles of 32) x (16 hidden
// tiles of 32), 128 threads (4 waves). Wave w handles batch sub-tile (w>>1)
// and hidden 16-tile (w&1): accumulates i/f/g/o gate tiles (K=1024) and the
// two field-gate tiles (K=128) with v_wmma_f32_16x16x32_bf16, then does the
// LSTM cell update in registers. Steps separated by grid_barrier.
// ---------------------------------------------------------------------------
#define AST 1032  // LDS activation row stride in shorts (1024 + 8 pad)
#define FST 136   // LDS field row stride in shorts (128 + 8 pad)
#define SMEM_BYTES ((32 * AST + 32 * FST) * 2)

__launch_bounds__(128)
__global__ void lstm_persistent(const float* __restrict__ word,
                                const float* __restrict__ field,
                                const float* __restrict__ bias_f,
                                const float* __restrict__ biasz,
                                const unsigned short* __restrict__ WsG,
                                const unsigned short* __restrict__ WsZ,
                                float* __restrict__ h0,
                                float* __restrict__ h1,
                                float* __restrict__ c_st,
                                float* __restrict__ out_seq,
                                float* __restrict__ out_hT,
                                float* __restrict__ out_cT,
                                unsigned* __restrict__ bar) {
  extern __shared__ unsigned short smem[];
  unsigned short* act = smem;             // [32][AST] bf16: word_t || h_prev
  unsigned short* fld = smem + 32 * AST;  // [32][FST] bf16: field_t

  const int tid  = threadIdx.x;
  const int lane = tid & 31;
  const int wave = tid >> 5;
  const int b0   = blockIdx.x * 32;
  const int nh0  = blockIdx.y * 32 + (wave & 1) * 16;
  const int brow = (wave >> 1) * 16 + (lane & 15);  // LDS batch row
  const int hi   = lane >> 4;
  const int ntb  = nh0 >> 4;                        // [0,32)
  const int nblocks = gridDim.x * gridDim.y;

  // t-invariant weight fragment base pointers (per-lane)
  const unsigned short* aB0 = WsG + ((size_t)((0 * 32 + ntb) * 32) * 512) + lane * 16;
  const unsigned short* aB1 = WsG + ((size_t)((1 * 32 + ntb) * 32) * 512) + lane * 16;
  const unsigned short* aB2 = WsG + ((size_t)((2 * 32 + ntb) * 32) * 512) + lane * 16;
  const unsigned short* aB3 = WsG + ((size_t)((3 * 32 + ntb) * 32) * 512) + lane * 16;
  const unsigned short* zB0 = WsZ + ((size_t)(ntb * 4) * 512) + lane * 16;
  const unsigned short* zB1 = WsZ + ((size_t)((32 + ntb) * 4) * 512) + lane * 16;

  for (int t = 0; t < SQ; ++t) {
    // Opaque memory clobber: prevents LICM from hoisting the (loop-invariant)
    // weight-fragment loads out of the t-loop. Without this the compiler
    // preloads ~4KB/lane of weights, overflows the register file and turns
    // the inner loop into scratch-spill reloads (seen in round-2 asm).
    asm volatile("" ::: "memory");

    const float* hin = (t & 1) ? h1 : h0;
    float*       hout = (t & 1) ? h0 : h1;

    // ---- stage activations: 32 rows x 1024 floats -> bf16 ----
    for (int idx = tid; idx < 32 * 256; idx += 128) {
      int row = idx >> 8;
      int k4  = (idx & 255) * 4;
      const float* src = (k4 < INW)
          ? (word + ((size_t)(b0 + row) * SQ + t) * INW + k4)
          : (hin + (size_t)(b0 + row) * HS + (k4 - INW));
      float4 v = *reinterpret_cast<const float4*>(src);
      unsigned short* dst = &act[row * AST + k4];
      dst[0] = f32_to_bf16(v.x); dst[1] = f32_to_bf16(v.y);
      dst[2] = f32_to_bf16(v.z); dst[3] = f32_to_bf16(v.w);
    }
    for (int idx = tid; idx < 32 * 32; idx += 128) {
      int row = idx >> 5;
      int k4  = (idx & 31) * 4;
      float4 v = *reinterpret_cast<const float4*>(
          field + ((size_t)(b0 + row) * SQ + t) * FS + k4);
      unsigned short* dst = &fld[row * FST + k4];
      dst[0] = f32_to_bf16(v.x); dst[1] = f32_to_bf16(v.y);
      dst[2] = f32_to_bf16(v.z); dst[3] = f32_to_bf16(v.w);
    }
    __syncthreads();

    v8f acc[4] = {v8f{}, v8f{}, v8f{}, v8f{}};
    v8f az0 = {}, az1 = {};

#pragma unroll
    for (int kk = 0; kk < 32; ++kk) {
      // B fragment (activations): two 8-bf16 runs per lane from LDS (k = 8*hi.., 16+8*hi..)
      BF16Frag bfrag;
      const uint4* bsrc =
          reinterpret_cast<const uint4*>(&act[brow * AST + kk * 32 + 8 * hi]);
      bfrag.q[0] = bsrc[0]; bfrag.q[1] = bsrc[2];

      BF16Frag a0, a1, a2, a3;
      const uint4* p0 = reinterpret_cast<const uint4*>(aB0 + kk * 512);
      const uint4* p1 = reinterpret_cast<const uint4*>(aB1 + kk * 512);
      const uint4* p2 = reinterpret_cast<const uint4*>(aB2 + kk * 512);
      const uint4* p3 = reinterpret_cast<const uint4*>(aB3 + kk * 512);
      a0.q[0] = p0[0]; a0.q[1] = p0[1];
      a1.q[0] = p1[0]; a1.q[1] = p1[1];
      a2.q[0] = p2[0]; a2.q[1] = p2[1];
      a3.q[0] = p3[0]; a3.q[1] = p3[1];

      acc[0] = __builtin_amdgcn_wmma_f32_16x16x32_bf16(
          false, a0.v, false, bfrag.v, (short)0, acc[0], false, false);
      acc[1] = __builtin_amdgcn_wmma_f32_16x16x32_bf16(
          false, a1.v, false, bfrag.v, (short)0, acc[1], false, false);
      acc[2] = __builtin_amdgcn_wmma_f32_16x16x32_bf16(
          false, a2.v, false, bfrag.v, (short)0, acc[2], false, false);
      acc[3] = __builtin_amdgcn_wmma_f32_16x16x32_bf16(
          false, a3.v, false, bfrag.v, (short)0, acc[3], false, false);

      if (kk < 4) {  // field-gate path, K = 128 (statically resolved by unroll)
        BF16Frag ffrag;
        const uint4* fsrc =
            reinterpret_cast<const uint4*>(&fld[brow * FST + kk * 32 + 8 * hi]);
        ffrag.q[0] = fsrc[0]; ffrag.q[1] = fsrc[2];

        BF16Frag za, zb;
        const uint4* z0 = reinterpret_cast<const uint4*>(zB0 + kk * 512);
        const uint4* z1 = reinterpret_cast<const uint4*>(zB1 + kk * 512);
        za.q[0] = z0[0]; za.q[1] = z0[1];
        zb.q[0] = z1[0]; zb.q[1] = z1[1];
        az0 = __builtin_amdgcn_wmma_f32_16x16x32_bf16(
            false, za.v, false, ffrag.v, (short)0, az0, false, false);
        az1 = __builtin_amdgcn_wmma_f32_16x16x32_bf16(
            false, zb.v, false, ffrag.v, (short)0, az1, false, false);
      }
    }

    // ---- LSTM cell update; all gates for (b, nh) live in this lane ----
    const int b    = b0 + brow;
    const int ncol = nh0 + 8 * hi;  // D row r -> hidden ncol + r

#pragma unroll
    for (int r = 0; r < 8; ++r) {
      int nh = ncol + r;
      float gi = sigmoidf(acc[0][r] + bias_f[0 * HS + nh]);
      float gf = sigmoidf(acc[1][r] + bias_f[1 * HS + nh] + 1.0f);
      float gg = tanhf   (acc[2][r] + bias_f[2 * HS + nh]);
      float go = sigmoidf(acc[3][r] + bias_f[3 * HS + nh]);
      float lt = sigmoidf(az0[r] + biasz[nh]);
      float zt = tanhf   (az1[r] + biasz[HS + nh]);

      size_t si = (size_t)b * HS + nh;
      float cn = gf * c_st[si] + gi * gg + lt * zt;
      float hn = go * tanhf(cn);

      c_st[si] = cn;
      hout[si] = hn;
      out_seq[((size_t)b * SQ + t) * HS + nh] = hn;
      if (t == SQ - 1) { out_hT[si] = hn; out_cT[si] = cn; }
    }

    if (t != SQ - 1) grid_barrier(bar, (unsigned)(t + 1) * (unsigned)nblocks);
  }
}

// ---------------------------------------------------------------------------
extern "C" void kernel_launch(void* const* d_in, const int* in_sizes, int n_in,
                              void* d_out, int out_size, void* d_ws, size_t ws_size,
                              hipStream_t stream) {
  (void)in_sizes; (void)n_in; (void)out_size; (void)ws_size;
  const float* word   = (const float*)d_in[0];
  const float* field  = (const float*)d_in[1];
  const float* Wf     = (const float*)d_in[2];
  const float* Wz     = (const float*)d_in[3];
  const float* Uf     = (const float*)d_in[4];
  const float* bias_f = (const float*)d_in[5];
  const float* biasz  = (const float*)d_in[6];

  char* ws = (char*)d_ws;
  unsigned short* WsG = (unsigned short*)ws;                              // 4 MB
  unsigned short* WsZ = (unsigned short*)(ws + (size_t)4 * 1024 * 1024);  // 256 KB
  float* h0 = (float*)(ws + (size_t)4 * 1024 * 1024 + 256 * 1024);        // 256 KB
  float* h1 = h0 + (size_t)NB * HS;                                       // 256 KB
  float* c  = h1 + (size_t)NB * HS;                                       // 256 KB
  unsigned* bar = (unsigned*)(c + (size_t)NB * HS);                       // 4 B

  float* out_seq = (float*)d_out;
  float* out_hT  = out_seq + (size_t)NB * SQ * HS;
  float* out_cT  = out_hT + (size_t)NB * HS;

  int total = 128 * 32 * 512 + 64 * 4 * 512;
  prep_weights<<<(total + 255) / 256, 256, 0, stream>>>(Wf, Uf, Wz, WsG, WsZ);
  init_state<<<(NB * HS + 255) / 256, 256, 0, stream>>>(h0, c, bar);

  lstm_persistent<<<dim3(NB / 32, HS / 32), 128, SMEM_BYTES, stream>>>(
      word, field, bias_f, biasz, WsG, WsZ, h0, h1, c,
      out_seq, out_hT, out_cT, bar);
}
